// Diffusion_48318382080167
// MI455X (gfx1250) — matrix operand order repaired and checked by reading, not compiled, not run on hardware
//
#include <hip/hip_runtime.h>
#include <hip/hip_bf16.h>

typedef __attribute__((ext_vector_type(16))) __bf16 v16bf;
typedef __attribute__((ext_vector_type(8)))  __bf16 v8bf;
typedef __attribute__((ext_vector_type(8)))  float  v8f;

#define F_LEN 512
#define BATCH 128
#define NN    256
#define NS    3

// ---- WMMA fragment loaders (wave32, 16x16x32 bf16 layouts per CDNA5 ISA) ----

// A-matrix 16x32: lane l (0-15): M=l, V0-3 K=0..7, V4-7 K=16..23
//                 lane l (16-31): M=l-16, V0-3 K=8..15, V4-7 K=24..31
static __device__ inline v16bf load_a_frag(const __bf16* __restrict__ base, int ld,
                                           int m0, int k0, int lane) {
    int m    = m0 + (lane & 15);
    int half = lane >> 4;
    const __bf16* row = base + (long)m * ld;
    v8bf lo = *(const v8bf*)(row + k0 + half * 8);        // K = k0+half*8 .. +7
    v8bf hi = *(const v8bf*)(row + k0 + 16 + half * 8);   // K = k0+16+half*8 .. +7
    v16bf a;
#pragma unroll
    for (int i = 0; i < 8; ++i) { a[i] = lo[i]; a[i + 8] = hi[i]; }
    return a;
}

// B-matrix 32x16 with B[k][n] = M[n][k] (transpose access; M symmetric or genuinely
// transposed source). Lane l: N=l&15, elements 0-7 K = k0 + (l>>4)*8 + h, elements 8-15 K = k0 + 16 + (l>>4)*8 + (h-8).
static __device__ inline v16bf load_b_frag_tr(const __bf16* __restrict__ base, int ld,
                                              int n0, int k0, int lane) {
    int n    = n0 + (lane & 15);
    int half = lane >> 4;
    const __bf16* p = base + (long)n * ld + k0 + half * 8;
    v8bf lo = *(const v8bf*)(p);
    v8bf hi = *(const v8bf*)(p + 16);
    v16bf b;
#pragma unroll
    for (int i = 0; i < 8; ++i) { b[i] = lo[i]; b[i + 8] = hi[i]; }
    return b;
}

#define WMMA_BF16(A, B, C) \
    __builtin_amdgcn_wmma_f32_16x16x32_bf16(false, (A), false, (B), (short)0, (C), false, false)

// ---- Kernel 1: x (F,B,N) f32 -> xbf (B,N,F) bf16 + sq[b][n] = sum_f x^2 ----
__global__ void k_pack(const float* __restrict__ x, __bf16* __restrict__ xbf,
                       float* __restrict__ sq) {
    int b = blockIdx.x;
    int n = threadIdx.x;                 // 256 threads, coalesced over n
    const float* xp = x + (long)b * NN + n;          // x[f][b][n], f-stride = B*N
    __bf16* dst = xbf + ((long)b * NN + n) * F_LEN;  // per-thread contiguous row
    float acc = 0.f;
    for (int f = 0; f < F_LEN; f += 8) {
        v8bf v;
#pragma unroll
        for (int j = 0; j < 8; ++j) {
            float t = xp[(long)(f + j) * (BATCH * NN)];
            acc += t * t;
            v[j] = (__bf16)t;
        }
        *(v8bf*)(dst + f) = v;           // 16B store
    }
    sq[b * NN + n] = acc;
}

// ---- Kernel 2: Gram + dis + ka_s. One wave per 32x32 tile (2x2 reg blocking). ----
__global__ void k_gram(const __bf16* __restrict__ xbf, const float* __restrict__ sq,
                       const float* __restrict__ theta,
                       float* __restrict__ dis, __bf16* __restrict__ ka) {
    int wid = threadIdx.x >> 5, lane = threadIdx.x & 31;
    int t    = blockIdx.x * 4 + wid;     // [0, B*64)
    int b    = t >> 6;
    int tile = t & 63;
    int tm = tile >> 3, tn = tile & 7;   // 8x8 grid of 32x32 tiles

    const __bf16* X = xbf + (long)b * NN * F_LEN;
    v8f c00 = {}, c01 = {}, c10 = {}, c11 = {};
    for (int k0 = 0; k0 < F_LEN; k0 += 32) {
        v16bf a0 = load_a_frag(X, F_LEN, tm * 32,      k0, lane);
        v16bf a1 = load_a_frag(X, F_LEN, tm * 32 + 16, k0, lane);
        v16bf b0 = load_b_frag_tr(X, F_LEN, tn * 32,      k0, lane);
        v16bf b1 = load_b_frag_tr(X, F_LEN, tn * 32 + 16, k0, lane);
        c00 = WMMA_BF16(a0, b0, c00);
        c01 = WMMA_BF16(a0, b1, c01);
        c10 = WMMA_BF16(a1, b0, c10);
        c11 = WMMA_BF16(a1, b1, c11);
    }
    v8f cs[4] = {c00, c01, c10, c11};

    int half = lane >> 4;
    float e0 = __expf(theta[0]), e1 = __expf(theta[1]), e2 = __expf(theta[2]);
    const float inv_f2 = 1.0f / ((float)F_LEN * (float)F_LEN);
#pragma unroll
    for (int i = 0; i < 2; ++i) {
#pragma unroll
        for (int j = 0; j < 2; ++j) {
            v8f c = cs[i * 2 + j];
            int n = tn * 32 + j * 16 + (lane & 15);
            float sqn = sq[b * NN + n];
#pragma unroll
            for (int r = 0; r < 8; ++r) {
                int m = tm * 32 + i * 16 + r + 8 * half;
                float d2 = fmaxf(sq[b * NN + m] + sqn - 2.0f * c[r], 0.0f);
                float ds = sqrtf(d2);
                float dm = d2 * inv_f2;      // (dis/F)^2 == max(d2,0)/F^2
                long o = ((long)b * NN + m) * NN + n;
                dis[o] = ds;
                long kb = (long)b * NS * NN * NN + (long)m * NN + n;
                ka[kb]                 = (__bf16)__expf(-e0 * dm);
                ka[kb + (long)NN * NN] = (__bf16)__expf(-e1 * dm);
                ka[kb + 2L * NN * NN]  = (__bf16)__expf(-e2 * dm);
            }
        }
    }
}

// ---- Kernel 3: a2 = ka @ ka / N  (ka symmetric => B operand is a row read) ----
__global__ void k_a2(const __bf16* __restrict__ ka, __bf16* __restrict__ a2) {
    int wid = threadIdx.x >> 5, lane = threadIdx.x & 31;
    int t    = blockIdx.x * 4 + wid;     // [0, B*S*64)
    int bs   = t >> 6;
    int tile = t & 63;
    int tm = tile >> 3, tn = tile & 7;

    const __bf16* A = ka + (long)bs * NN * NN;
    v8f c00 = {}, c01 = {}, c10 = {}, c11 = {};
    for (int k0 = 0; k0 < NN; k0 += 32) {
        v16bf a0 = load_a_frag(A, NN, tm * 32,      k0, lane);
        v16bf a1 = load_a_frag(A, NN, tm * 32 + 16, k0, lane);
        v16bf b0 = load_b_frag_tr(A, NN, tn * 32,      k0, lane);  // ka symmetric
        v16bf b1 = load_b_frag_tr(A, NN, tn * 32 + 16, k0, lane);
        c00 = WMMA_BF16(a0, b0, c00);
        c01 = WMMA_BF16(a0, b1, c01);
        c10 = WMMA_BF16(a1, b0, c10);
        c11 = WMMA_BF16(a1, b1, c11);
    }
    v8f cs[4] = {c00, c01, c10, c11};

    const float invn = 1.0f / (float)NN;
    int half = lane >> 4;
    __bf16* D = a2 + (long)bs * NN * NN;
#pragma unroll
    for (int i = 0; i < 2; ++i) {
#pragma unroll
        for (int j = 0; j < 2; ++j) {
            v8f c = cs[i * 2 + j];
            int n = tn * 32 + j * 16 + (lane & 15);
#pragma unroll
            for (int r = 0; r < 8; ++r) {
                int m = tm * 32 + i * 16 + r + 8 * half;
                D[(long)m * NN + n] = (__bf16)(c[r] * invn);
            }
        }
    }
}

// ---- Kernel 4: a3 GEMM (32x32 tile) + full fused epilogue -> out ----
__global__ void k_final(const __bf16* __restrict__ ka, const __bf16* __restrict__ a2,
                        const float* __restrict__ dis,
                        const float* __restrict__ alphas, const float* __restrict__ att0,
                        const float* __restrict__ dfw2, const float* __restrict__ dfw3,
                        const float* __restrict__ conv_w, const float* __restrict__ conv_b,
                        const float* __restrict__ bn_g, const float* __restrict__ bn_b,
                        const float* __restrict__ bn_m, const float* __restrict__ bn_v,
                        float* __restrict__ out) {
    int wid = threadIdx.x >> 5, lane = threadIdx.x & 31;
    int t    = blockIdx.x * 4 + wid;     // [0, B*64)
    int b    = t >> 6;
    int tile = t & 63;
    int tm = tile >> 3, tn = tile & 7;
    int half = lane >> 4;

    float w2 = 0.3f * dfw2[0], w3 = 0.3f * dfw3[0];
    float y[4][8];
#pragma unroll
    for (int q = 0; q < 4; ++q)
#pragma unroll
        for (int r = 0; r < 8; ++r) y[q][r] = conv_b[0];

    for (int s = 0; s < NS; ++s) {
        const __bf16* Ap = a2 + ((long)b * NS + s) * NN * NN;
        const __bf16* Kp = ka + ((long)b * NS + s) * NN * NN;
        v8f c00 = {}, c01 = {}, c10 = {}, c11 = {};
        for (int k0 = 0; k0 < NN; k0 += 32) {
            v16bf a0 = load_a_frag(Ap, NN, tm * 32,      k0, lane);     // a3 = a2 @ ka
            v16bf a1 = load_a_frag(Ap, NN, tm * 32 + 16, k0, lane);
            v16bf b0 = load_b_frag_tr(Kp, NN, tn * 32,      k0, lane);  // ka symmetric
            v16bf b1 = load_b_frag_tr(Kp, NN, tn * 32 + 16, k0, lane);
            c00 = WMMA_BF16(a0, b0, c00);
            c01 = WMMA_BF16(a0, b1, c01);
            c10 = WMMA_BF16(a1, b0, c10);
            c11 = WMMA_BF16(a1, b1, c11);
        }
        v8f cs[4] = {c00, c01, c10, c11};

        float al = alphas[s], cw = conv_w[s];
        const float* a0p = att0 + (long)s * NN * NN;
#pragma unroll
        for (int i = 0; i < 2; ++i) {
#pragma unroll
            for (int j = 0; j < 2; ++j) {
                v8f c = cs[i * 2 + j];
                int n = tn * 32 + j * 16 + (lane & 15);
#pragma unroll
                for (int r = 0; r < 8; ++r) {
                    int m = tm * 32 + i * 16 + r + 8 * half;
                    long o = (long)m * NN + n;
                    float kav = (float)Kp[o];
                    float a2v = (float)Ap[o];
                    float a3v = c[r] * (1.0f / (float)NN);
                    float v   = 0.4f * kav + w2 * a2v + w3 * a3v;
                    float att = tanhf(v) * al + a0p[o];
                    y[i * 2 + j][r] += cw * att;
                }
            }
        }
    }

    float scale = bn_g[0] * rsqrtf(bn_v[0] + 1e-5f);
    float bmean = bn_m[0], bbeta = bn_b[0];
    const float* Db = dis + (long)b * NN * NN;
    float* Ob = out + (long)b * NN * NN;
#pragma unroll
    for (int i = 0; i < 2; ++i) {
#pragma unroll
        for (int j = 0; j < 2; ++j) {
            int n = tn * 32 + j * 16 + (lane & 15);
#pragma unroll
            for (int r = 0; r < 8; ++r) {
                int m = tm * 32 + i * 16 + r + 8 * half;
                long o = (long)m * NN + n;
                float yy = (y[i * 2 + j][r] - bmean) * scale + bbeta;
                float z = Db[o] + yy;
                Ob[o] = z > 0.f ? z : 0.1f * z;   // LeakyReLU(0.1)
            }
        }
    }
}

extern "C" void kernel_launch(void* const* d_in, const int* in_sizes, int n_in,
                              void* d_out, int out_size, void* d_ws, size_t ws_size,
                              hipStream_t stream) {
    const float* x      = (const float*)d_in[0];
    const float* theta  = (const float*)d_in[1];
    const float* alphas = (const float*)d_in[2];
    const float* att0   = (const float*)d_in[3];
    const float* dfw2   = (const float*)d_in[4];
    const float* dfw3   = (const float*)d_in[5];
    const float* conv_w = (const float*)d_in[6];
    const float* conv_b = (const float*)d_in[7];
    const float* bn_g   = (const float*)d_in[8];
    const float* bn_b   = (const float*)d_in[9];
    const float* bn_m   = (const float*)d_in[10];
    const float* bn_v   = (const float*)d_in[11];

    // workspace layout (bytes):
    //   xbf : [0,            33554432)   B*N*F bf16
    //   ka  : [33554432,     83886080)   B*S*N*N bf16
    //   a2  : [83886080,    134217728)   B*S*N*N bf16
    //   dis : [134217728,   167772160)   B*N*N f32
    //   sq  : [167772160,   167903232)   B*N f32
    char* ws = (char*)d_ws;
    __bf16* xbf = (__bf16*)(ws);
    __bf16* ka  = (__bf16*)(ws + 33554432);
    __bf16* a2  = (__bf16*)(ws + 83886080);
    float*  dis = (float*)(ws + 134217728);
    float*  sq  = (float*)(ws + 167772160);
    float*  out = (float*)d_out;

    k_pack <<<BATCH,                 256, 0, stream>>>(x, xbf, sq);
    k_gram <<<(BATCH * 64) / 4,      128, 0, stream>>>(xbf, sq, theta, dis, ka);
    k_a2   <<<(BATCH * NS * 64) / 4, 128, 0, stream>>>(ka, a2);
    k_final<<<(BATCH * 64) / 4,      128, 0, stream>>>(ka, a2, dis, alphas, att0,
                                                       dfw2, dfw3, conv_w, conv_b,
                                                       bn_g, bn_b, bn_m, bn_v, out);
}
